// MDM_86947317940747
// MI455X (gfx1250) — matrix operand order repaired and checked by reading, not compiled, not run on hardware
//
#include <hip/hip_runtime.h>
#include <hip/hip_bf16.h>
#include <math.h>

// ---------------- problem constants ----------------
#define S_   1024
#define B_   2
#define D_   1024
#define H_   16
#define DH_  64
#define L_   4
#define WIN_ 64
#define SB_  (S_*B_)     // 2048 rows in [S,B,D] flattened as [(s*B+b), D]

typedef __attribute__((ext_vector_type(16))) __bf16         v16bf;
typedef __attribute__((ext_vector_type(8)))  float          v8f;
typedef __attribute__((ext_vector_type(8)))  unsigned short ushort8;

// float -> bf16 (round-to-nearest-even), kept as raw ushort for storage
static __device__ __forceinline__ unsigned short f2bfu(float f) {
    unsigned u = __builtin_bit_cast(unsigned, f);
    u += 0x7fffu + ((u >> 16) & 1u);
    return (unsigned short)(u >> 16);
}
static __device__ __forceinline__ __bf16 u2bf(unsigned short h) {
    return __builtin_bit_cast(__bf16, h);
}

// A-fragment (16x32 bf16): each lane reads 16 contiguous elements along K:
// K = kbase+[0..15].
// 'base' must already point at row*ld + k0 + kbase.
static __device__ __forceinline__ v16bf load_afrag(const unsigned short* base) {
    ushort8 lo = *reinterpret_cast<const ushort8*>(base);
    ushort8 hi = *reinterpret_cast<const ushort8*>(base + 8);
    v16bf a;
#pragma unroll
    for (int i = 0; i < 8; ++i) { a[i] = u2bf(lo[i]); a[8 + i] = u2bf(hi[i]); }
    return a;
}

// B-fragment (32x16 bf16) from a [N,K] row-major matrix (B[k][n] = W[n][k]):
// each lane reads 16 contiguous elements along K (32B).
// 'base' must point at n*ld + k0 + koff (koff = 0 or 16 by lane half).
static __device__ __forceinline__ v16bf load_bfrag(const unsigned short* base) {
    ushort8 lo = *reinterpret_cast<const ushort8*>(base);
    ushort8 hi = *reinterpret_cast<const ushort8*>(base + 8);
    v16bf b;
#pragma unroll
    for (int i = 0; i < 8; ++i) { b[i] = u2bf(lo[i]); b[8 + i] = u2bf(hi[i]); }
    return b;
}

static __device__ __forceinline__ v8f zero8() {
    v8f z;
#pragma unroll
    for (int i = 0; i < 8; ++i) z[i] = 0.0f;
    return z;
}

// ---------------- fp32 -> bf16 weight conversion ----------------
__global__ __launch_bounds__(256) void cvt_bf16_k(const float* __restrict__ src,
                                                  unsigned short* __restrict__ dst,
                                                  size_t n) {
    size_t i = (size_t)blockIdx.x * 256 + threadIdx.x;
    if (i < n) dst[i] = f2bfu(src[i]);
}

// ---------------- adaLN modulation: mod[l,b,:] = relu(c[b]) @ wada[l]^T + bada[l]
__global__ __launch_bounds__(256) void adamod_k(const float* __restrict__ c,
                                                const float* __restrict__ wada,
                                                const float* __restrict__ bada,
                                                float* __restrict__ mod) {
    int idx = blockIdx.x * 256 + threadIdx.x;      // (l*B + b)*6D + j
    int j  = idx % (6 * D_);
    int lb = idx / (6 * D_);
    int b  = lb % B_;
    int l  = lb / B_;
    const float* w  = wada + ((size_t)l * 6 * D_ + j) * D_;
    const float* cc = c + (size_t)b * D_;
    float a = bada[(size_t)l * 6 * D_ + j];
    for (int k = 0; k < D_; ++k) {
        float cv = cc[k];
        a += (cv > 0.0f ? cv : 0.0f) * w[k];
    }
    mod[idx] = a;
}

// ---------------- modulated LayerNorm: xm = LN(x)*(1+sc) + sh, bf16 out ----
__global__ __launch_bounds__(256) void ln_mod_k(const float* __restrict__ x,
                                                const float* __restrict__ modl, // [B,6D]
                                                int shOff, int scOff,
                                                unsigned short* __restrict__ xm) {
    __shared__ float red[256];
    int row = blockIdx.x;
    int b   = row & (B_ - 1);
    const float* xr = x + (size_t)row * D_;
    int d0 = threadIdx.x * 4;
    float v0 = xr[d0], v1 = xr[d0 + 1], v2 = xr[d0 + 2], v3 = xr[d0 + 3];
    float s  = v0 + v1 + v2 + v3;
    float ss = v0 * v0 + v1 * v1 + v2 * v2 + v3 * v3;

    red[threadIdx.x] = s; __syncthreads();
    for (int o = 128; o > 0; o >>= 1) {
        if (threadIdx.x < o) red[threadIdx.x] += red[threadIdx.x + o];
        __syncthreads();
    }
    float mean = red[0] * (1.0f / D_);
    __syncthreads();
    red[threadIdx.x] = ss; __syncthreads();
    for (int o = 128; o > 0; o >>= 1) {
        if (threadIdx.x < o) red[threadIdx.x] += red[threadIdx.x + o];
        __syncthreads();
    }
    float var = red[0] * (1.0f / D_) - mean * mean;
    float rs  = rsqrtf(var + 1e-6f);

    const float* sh = modl + (size_t)b * 6 * D_ + shOff;
    const float* sc = modl + (size_t)b * 6 * D_ + scOff;
    unsigned short* o = xm + (size_t)row * D_ + d0;
    o[0] = f2bfu((v0 - mean) * rs * (1.0f + sc[d0 + 0]) + sh[d0 + 0]);
    o[1] = f2bfu((v1 - mean) * rs * (1.0f + sc[d0 + 1]) + sh[d0 + 1]);
    o[2] = f2bfu((v2 - mean) * rs * (1.0f + sc[d0 + 2]) + sh[d0 + 2]);
    o[3] = f2bfu((v3 - mean) * rs * (1.0f + sc[d0 + 3]) + sh[d0 + 3]);
}

// ---------------- generic WMMA GEMM: out = A[M,K](bf16) @ W[N,K]^T + bias ----
// Per wave: 64x32 output tile (8 WMMAs per 32-wide k-step). Ping-pong
// register double-buffering with a manual 2x unroll: no rotation moves, so
// loads for chunk k+1 overlap the 8 WMMAs of chunk k without v_mov/v_nop
// overhead. Requires (K/32) even (holds: K = 1024 or 4096).
// epi 0: store bf16;  epi 1: store relu bf16;
// epi 2: xout = xin + gate[b,gOff+n] * (acc + bias)   (fused gated residual)
struct Frags {
    v16bf bf0, bf1;
    v16bf af[4];
};

static __device__ __forceinline__ void load_set(Frags& f,
                                                const unsigned short* bptr0,
                                                const unsigned short* bptr1,
                                                const unsigned short* aptr,
                                                int k0, size_t K) {
    f.bf0 = load_bfrag(bptr0 + k0);
    f.bf1 = load_bfrag(bptr1 + k0);
#pragma unroll
    for (int t = 0; t < 4; ++t)
        f.af[t] = load_afrag(aptr + (size_t)t * 16 * K + k0);
}

static __device__ __forceinline__ void mma_set(v8f acc[2][4], const Frags& f) {
#pragma unroll
    for (int t = 0; t < 4; ++t) {
        acc[0][t] = __builtin_amdgcn_wmma_f32_16x16x32_bf16(
                        false, f.af[t], false, f.bf0, (short)0, acc[0][t], false, false);
        acc[1][t] = __builtin_amdgcn_wmma_f32_16x16x32_bf16(
                        false, f.af[t], false, f.bf1, (short)0, acc[1][t], false, false);
    }
}

__global__ __launch_bounds__(256) void gemm_bf16_k(
        const unsigned short* __restrict__ A,
        const unsigned short* __restrict__ Bw,
        const float* __restrict__ bias,
        int M, int N, int K, int epi,
        const float* __restrict__ gate, int gOff,
        const float* __restrict__ xin, float* __restrict__ xout,
        unsigned short* __restrict__ outb) {
    int lane = threadIdx.x & 31;
    int wave = threadIdx.x >> 5;
    int n0   = blockIdx.x * 256 + wave * 32;
    int m0   = blockIdx.y * 64;
    int l15  = lane & 15;
    int kba  = (lane >> 4) << 4;    // A fragment k-base: 0 / 16
    int kbb  = (lane >> 4) << 4;    // B fragment k-base: 0 / 16

    v8f acc[2][4];
#pragma unroll
    for (int j = 0; j < 2; ++j)
#pragma unroll
        for (int t = 0; t < 4; ++t) acc[j][t] = zero8();

    const unsigned short* bptr0 = Bw + (size_t)(n0 + l15) * K + kbb;
    const unsigned short* bptr1 = bptr0 + (size_t)16 * K;
    const unsigned short* aptr  = A  + (size_t)(m0 + l15) * K + kba;
    size_t Ks = (size_t)K;

    Frags fr0, fr1;
    load_set(fr0, bptr0, bptr1, aptr, 0, Ks);      // chunk 0

    int k0 = 32;
    for (; k0 + 32 < K; k0 += 64) {
        load_set(fr1, bptr0, bptr1, aptr, k0, Ks);         // chunk i+1
        __builtin_prefetch(bptr0 + k0 + 512, 0, 1);
        mma_set(acc, fr0);                                  // chunk i
        load_set(fr0, bptr0, bptr1, aptr, k0 + 32, Ks);    // chunk i+2
        mma_set(acc, fr1);                                  // chunk i+1
    }
    load_set(fr1, bptr0, bptr1, aptr, K - 32, Ks);         // last chunk
    mma_set(acc, fr0);
    mma_set(acc, fr1);

    int mo = (lane >> 4) << 3;        // C/D row offset by lane half
#pragma unroll
    for (int j = 0; j < 2; ++j) {
        int n = n0 + j * 16 + l15;
        float bv = bias[n];
#pragma unroll
        for (int t = 0; t < 4; ++t) {
#pragma unroll
            for (int r = 0; r < 8; ++r) {
                int m = m0 + t * 16 + r + mo;
                float v = acc[j][t][r] + bv;
                size_t o = (size_t)m * N + n;
                if (epi == 0) {
                    outb[o] = f2bfu(v);
                } else if (epi == 1) {
                    outb[o] = f2bfu(v > 0.0f ? v : 0.0f);
                } else {
                    int   bb = m & (B_ - 1);
                    float g  = gate[(size_t)bb * 6 * D_ + gOff + n];
                    xout[o]  = xin[o] + g * v;
                }
            }
        }
    }
}

// ---------------- banded attention: one wave per (16-query tile, b, h) -----
// scores = Q K^T / 8 masked to |s-t|<=64; softmax; O = P V ; all via WMMA bf16
__global__ __launch_bounds__(32) void attn_k(const unsigned short* __restrict__ qkv, // [SB,3D] bf16
                                             unsigned short* __restrict__ obuf) {    // [SB,D]  bf16
    __shared__ float          slds[16 * 160];
    __shared__ unsigned short plds[16 * 160];

    int lane = threadIdx.x;
    int qt = blockIdx.x, b = blockIdx.y, h = blockIdx.z;
    int qs  = qt * 16;
    int l15 = lane & 15;
    int kba = (lane >> 4) << 4;
    int kbb = (lane >> 4) << 4;
    const int KO = D_, VO = 2 * D_;
    const int LDQ = 3 * D_;

    // Q fragments (A, 16x32 over dh twice): row = qs + l15
    const unsigned short* qbase =
        qkv + ((size_t)(qs + l15) * B_ + b) * LDQ + h * DH_ + kba;
    v16bf aq0 = load_afrag(qbase);
    v16bf aq1 = load_afrag(qbase + 32);

    int kt0 = qt - 4; if (kt0 < 0) kt0 = 0;
    int kt1 = qt + 4; if (kt1 > S_ / 16 - 1) kt1 = S_ / 16 - 1;
    int nt  = kt1 - kt0 + 1;

    // scores: S16x16 = Q(16x64) K^T(64x16) per key tile, masked, into LDS
    for (int kt = kt0; kt <= kt1; ++kt) {
        const unsigned short* kb =
            qkv + ((size_t)((kt * 16 + l15) * B_ + b)) * LDQ + KO + h * DH_ + kbb;
        v16bf bk0 = load_bfrag(kb);
        v16bf bk1 = load_bfrag(kb + 32);
        v8f cfrag = zero8();
        cfrag = __builtin_amdgcn_wmma_f32_16x16x32_bf16(false, aq0, false, bk0,
                                                        (short)0, cfrag, false, false);
        cfrag = __builtin_amdgcn_wmma_f32_16x16x32_bf16(false, aq1, false, bk1,
                                                        (short)0, cfrag, false, false);
        int mo = (lane >> 4) << 3;
#pragma unroll
        for (int r = 0; r < 8; ++r) {
            int m = r + mo;
            int t = kt * 16 + l15;
            int dist = (qs + m) - t; if (dist < 0) dist = -dist;
            float v = cfrag[r] * 0.125f;
            if (dist > WIN_) v += -1e9f;
            slds[m * 160 + (kt - kt0) * 16 + l15] = v;
        }
    }
    __syncthreads();

    // softmax along the band (rows handled by lanes 0..15)
    int ncol  = nt * 16;
    int nt2   = (nt + 1) & ~1;          // pad to an even number of key tiles
    int ncol2 = nt2 * 16;
    if (lane < 16) {
        int r = lane;
        float mx = -1e30f;
        for (int j = 0; j < ncol; ++j) { float v = slds[r * 160 + j]; if (v > mx) mx = v; }
        float sum = 0.0f;
        for (int j = 0; j < ncol; ++j) sum += __expf(slds[r * 160 + j] - mx);
        float inv = 1.0f / sum;
        for (int j = 0; j < ncol; ++j)
            plds[r * 160 + j] = f2bfu(__expf(slds[r * 160 + j] - mx) * inv);
        for (int j = ncol; j < ncol2; ++j) plds[r * 160 + j] = 0;
    }
    __syncthreads();

    // O(16x64) = P(16xT) V(Tx64): K-dim in 32-key steps, 4 dh tiles of 16
    v8f acc[4];
#pragma unroll
    for (int t = 0; t < 4; ++t) acc[t] = zero8();
    int npair = nt2 >> 1;
    for (int pr = 0; pr < npair; ++pr) {
        const unsigned short* pbase = &plds[l15 * 160 + pr * 32 + kba];
        v16bf ap = load_afrag(pbase);
#pragma unroll
        for (int dt = 0; dt < 4; ++dt) {
            v16bf bv;
#pragma unroll
            for (int i = 0; i < 16; ++i) {
                int t = (kt0 + pr * 2) * 16 + kbb + i;
                unsigned short u = 0;
                if (t < S_)
                    u = qkv[((size_t)(t * B_ + b)) * LDQ + VO + h * DH_ + dt * 16 + l15];
                bv[i] = u2bf(u);
            }
            acc[dt] = __builtin_amdgcn_wmma_f32_16x16x32_bf16(
                          false, ap, false, bv, (short)0, acc[dt], false, false);
        }
    }

    int mo = (lane >> 4) << 3;
#pragma unroll
    for (int dt = 0; dt < 4; ++dt)
#pragma unroll
        for (int r = 0; r < 8; ++r) {
            int m = r + mo;
            obuf[((size_t)((qs + m) * B_ + b)) * D_ + h * DH_ + dt * 16 + l15] =
                f2bfu(acc[dt][r]);
        }
}

// ---------------- host orchestration ----------------
extern "C" void kernel_launch(void* const* d_in, const int* in_sizes, int n_in,
                              void* d_out, int out_size, void* d_ws, size_t ws_size,
                              hipStream_t stream) {
    (void)in_sizes; (void)n_in; (void)out_size; (void)ws_size;
    const float* x    = (const float*)d_in[0];
    const float* c    = (const float*)d_in[1];
    const float* wqkv = (const float*)d_in[2];
    const float* bqkv = (const float*)d_in[3];
    const float* wout = (const float*)d_in[4];
    const float* bout = (const float*)d_in[5];
    const float* w1   = (const float*)d_in[6];
    const float* b1   = (const float*)d_in[7];
    const float* w2   = (const float*)d_in[8];
    const float* b2   = (const float*)d_in[9];
    const float* wada = (const float*)d_in[10];
    const float* bada = (const float*)d_in[11];
    float* out = (float*)d_out;

    char* ws = (char*)d_ws;
    size_t off = 0;
    auto alloc = [&](size_t bytes) -> void* {
        void* p = ws + off;
        off += (bytes + 255) & ~(size_t)255;
        return p;
    };
    unsigned short* wqkvb = (unsigned short*)alloc((size_t)L_ * 3 * D_ * D_ * 2);
    unsigned short* woutb = (unsigned short*)alloc((size_t)L_ * D_ * D_ * 2);
    unsigned short* w1b   = (unsigned short*)alloc((size_t)L_ * 4 * D_ * D_ * 2);
    unsigned short* w2b   = (unsigned short*)alloc((size_t)L_ * D_ * 4 * D_ * 2);
    float*          xb    = (float*)alloc((size_t)SB_ * D_ * 4);
    unsigned short* xmb   = (unsigned short*)alloc((size_t)SB_ * D_ * 2);
    unsigned short* qkvb  = (unsigned short*)alloc((size_t)SB_ * 3 * D_ * 2);
    unsigned short* ob    = (unsigned short*)alloc((size_t)SB_ * D_ * 2);
    unsigned short* hb    = (unsigned short*)alloc((size_t)SB_ * 4 * D_ * 2);
    float*          modb  = (float*)alloc((size_t)L_ * B_ * 6 * D_ * 4);

    // one-time weight downconversion fp32 -> bf16 (bf16 set ~96MB: fits in L2)
    size_t n;
    n = (size_t)L_ * 3 * D_ * D_;
    cvt_bf16_k<<<(unsigned)((n + 255) / 256), 256, 0, stream>>>(wqkv, wqkvb, n);
    n = (size_t)L_ * D_ * D_;
    cvt_bf16_k<<<(unsigned)((n + 255) / 256), 256, 0, stream>>>(wout, woutb, n);
    n = (size_t)L_ * 4 * D_ * D_;
    cvt_bf16_k<<<(unsigned)((n + 255) / 256), 256, 0, stream>>>(w1, w1b, n);
    n = (size_t)L_ * D_ * 4 * D_;
    cvt_bf16_k<<<(unsigned)((n + 255) / 256), 256, 0, stream>>>(w2, w2b, n);

    adamod_k<<<(L_ * B_ * 6 * D_) / 256, 256, 0, stream>>>(c, wada, bada, modb);
    hipMemcpyAsync(xb, x, (size_t)SB_ * D_ * sizeof(float),
                   hipMemcpyDeviceToDevice, stream);

    for (int l = 0; l < L_; ++l) {
        const float* modl = modb + (size_t)l * B_ * 6 * D_;

        // attention branch
        ln_mod_k<<<SB_, 256, 0, stream>>>(xb, modl, /*sh*/0, /*sc*/D_, xmb);
        gemm_bf16_k<<<dim3(3 * D_ / 256, SB_ / 64), 256, 0, stream>>>(
            xmb, wqkvb + (size_t)l * 3 * D_ * D_, bqkv + (size_t)l * 3 * D_,
            SB_, 3 * D_, D_, /*epi*/0, nullptr, 0, nullptr, nullptr, qkvb);
        attn_k<<<dim3(S_ / 16, B_, H_), 32, 0, stream>>>(qkvb, ob);
        gemm_bf16_k<<<dim3(D_ / 256, SB_ / 64), 256, 0, stream>>>(
            ob, woutb + (size_t)l * D_ * D_, bout + (size_t)l * D_,
            SB_, D_, D_, /*epi*/2, modl, /*g_msa*/2 * D_, xb, xb, nullptr);

        // MLP branch
        ln_mod_k<<<SB_, 256, 0, stream>>>(xb, modl, /*sh*/3 * D_, /*sc*/4 * D_, xmb);
        gemm_bf16_k<<<dim3(4 * D_ / 256, SB_ / 64), 256, 0, stream>>>(
            xmb, w1b + (size_t)l * 4 * D_ * D_, b1 + (size_t)l * 4 * D_,
            SB_, 4 * D_, D_, /*epi*/1, nullptr, 0, nullptr, nullptr, hb);
        gemm_bf16_k<<<dim3(D_ / 256, SB_ / 64), 256, 0, stream>>>(
            hb, w2b + (size_t)l * D_ * 4 * D_, b2 + (size_t)l * D_,
            SB_, D_, 4 * D_, /*epi*/2, modl, /*g_mlp*/5 * D_, xb, xb, nullptr);
    }

    hipMemcpyAsync(out, xb, (size_t)SB_ * D_ * sizeof(float),
                   hipMemcpyDeviceToDevice, stream);
}
